// NGCF_56057913147466
// MI455X (gfx1250) — matrix operand order repaired and checked by reading, not compiled, not run on hardware
//
#include <hip/hip_runtime.h>

#define N_ITEM_ 50000
#define NN_     100000
#define EMB_    512
#define NNZ_    3200000
#define LAYERS_ 3
#define BATCH_  4096
#define LRELU_  0.2f
#define EPS_    1e-12f

typedef __attribute__((ext_vector_type(16))) __bf16  v16bf;
typedef __attribute__((ext_vector_type(8)))  float   v8f;

union FragU { v16bf v; uint4 q[2]; };

#if defined(__HIP_DEVICE_COMPILE__) && __has_builtin(__builtin_amdgcn_global_load_async_to_lds_b128)
#define HAS_ASYNC_LDS 1
#else
#define HAS_ASYNC_LDS 0
#endif

#if HAS_ASYNC_LDS
typedef int v4i_ __attribute__((vector_size(4 * sizeof(int))));
typedef __attribute__((address_space(1))) v4i_ gv4i_t;   // global 16B vector
typedef __attribute__((address_space(3))) v4i_ lv4i_t;   // LDS 16B vector

__device__ __forceinline__ void async_ld128(const void* g, void* l) {
  __builtin_amdgcn_global_load_async_to_lds_b128(
      (gv4i_t*)g, (lv4i_t*)l, /*offset=*/0, /*cpol=*/0);
}
__device__ __forceinline__ void wait_async0() {
#if __has_builtin(__builtin_amdgcn_s_wait_asynccnt)
  __builtin_amdgcn_s_wait_asynccnt(0);
#else
  asm volatile("s_wait_asynccnt 0x0" ::: "memory");
#endif
}
#endif

__device__ __forceinline__ unsigned short f2bf(float f) {
  unsigned u = __float_as_uint(f);
  u += 0x7FFFu + ((u >> 16) & 1u);           // round-to-nearest-even
  return (unsigned short)(u >> 16);
}
__device__ __forceinline__ unsigned pack_bf2(float lo, float hi) {
  return (unsigned)f2bf(lo) | ((unsigned)f2bf(hi) << 16);
}

// ---------------------------------------------------------------- init: ego = acc = concat(vision, text)
__global__ void init_ego_acc(const float* __restrict__ vis, const float* __restrict__ txt,
                             float* __restrict__ ego, float* __restrict__ acc) {
  size_t i4 = (size_t)blockIdx.x * blockDim.x + threadIdx.x;     // float4 index
  const size_t tot = (size_t)NN_ * (EMB_ / 4);
  if (i4 >= tot) return;
  size_t row = i4 >> 7;
  size_t c4  = i4 & 127;
  const float4* src = (row < N_ITEM_)
      ? ((const float4*)vis) + row * 128 + c4
      : ((const float4*)txt) + (row - N_ITEM_) * 128 + c4;
  float4 v = *src;
  ((float4*)ego)[i4] = v;
  ((float4*)acc)[i4] = v;
}

// ---------------------------------------------------------------- SpMM: out[row] += val * x[col], wave per edge
__global__ void spmm_atomic(const int* __restrict__ rows, const int* __restrict__ cols,
                            const float* __restrict__ vals,
                            const float* __restrict__ x0, const float* __restrict__ x1,
                            float* __restrict__ out) {
  const int e = blockIdx.x * 8 + (threadIdx.x >> 5);
  if (e >= NNZ_) return;
  const int l = threadIdx.x & 31;
  const int r = rows[e];
  const int c = cols[e];
  const float v = vals[e];
  const float* src = (c < N_ITEM_) ? (x0 + (size_t)c * EMB_)
                                   : (x1 + (size_t)(c - N_ITEM_) * EMB_);
  float* dst = out + (size_t)r * EMB_;
#pragma unroll
  for (int j = 0; j < 4; ++j) {
    const int idx = (l + j * 32) * 4;
    float4 xv = *(const float4*)(src + idx);
    __hip_atomic_fetch_add(dst + idx + 0, v * xv.x, __ATOMIC_RELAXED, __HIP_MEMORY_SCOPE_AGENT);
    __hip_atomic_fetch_add(dst + idx + 1, v * xv.y, __ATOMIC_RELAXED, __HIP_MEMORY_SCOPE_AGENT);
    __hip_atomic_fetch_add(dst + idx + 2, v * xv.z, __ATOMIC_RELAXED, __HIP_MEMORY_SCOPE_AGENT);
    __hip_atomic_fetch_add(dst + idx + 3, v * xv.w, __ATOMIC_RELAXED, __HIP_MEMORY_SCOPE_AGENT);
  }
}

// ---------------------------------------------------------------- pack bf16 A matrices: [side | edis | ego*side + oge*edis]
__global__ void build_abf(const float* __restrict__ side, const float* __restrict__ edis,
                          const float* __restrict__ ego,
                          const float* __restrict__ vis, const float* __restrict__ txt,
                          unsigned* __restrict__ a0, unsigned* __restrict__ a1,
                          unsigned* __restrict__ a2) {
  const size_t p = (size_t)blockIdx.x * blockDim.x + threadIdx.x;  // bf16-pair index
  const size_t tot = (size_t)NN_ * (EMB_ / 2);
  if (p >= tot) return;
  const size_t i   = p * 2;
  const size_t row = i >> 9;
  const size_t col = i & 511;
  const float* ogeRow = (row < N_ITEM_) ? (txt + row * EMB_)
                                        : (vis + (row - N_ITEM_) * EMB_);
  const float s0 = side[i], s1 = side[i + 1];
  const float d0 = edis[i], d1 = edis[i + 1];
  const float g0 = ego[i],  g1 = ego[i + 1];
  const float o0 = ogeRow[col], o1 = ogeRow[col + 1];
  a0[p] = pack_bf2(s0, s1);
  a1[p] = pack_bf2(d0, d1);
  a2[p] = pack_bf2(g0 * s0 + o0 * d0, g1 * s1 + o1 * d1);
}

// ---------------------------------------------------------------- weights -> bf16, transposed: WbT[n][kcat], kcat = seg*512 + kk
__global__ void convert_weights(const float* __restrict__ wgc, const float* __restrict__ wgc2,
                                const float* __restrict__ wbi, unsigned short* __restrict__ wbt) {
  const int t = blockIdx.x * blockDim.x + threadIdx.x;
  if (t >= 3 * EMB_ * EMB_) return;
  const int seg = t >> 18;                       // EMB_*EMB_ == 2^18
  const int rem = t & (EMB_ * EMB_ - 1);         // = kk*512 + n
  const int kk  = rem >> 9;
  const int n   = rem & 511;
  const float* W = (seg == 0) ? wgc : (seg == 1) ? wgc2 : wbi;
  wbt[(size_t)n * (3 * EMB_) + seg * EMB_ + kk] = f2bf(W[rem]);
}

// ---------------------------------------------------------------- WMMA GEMM: Z[NN,512] = Abf[NN,1536] @ WbT^T + bias
#define BM  128
#define BN  128
#define BK  32
#define LDT 40   // padded LDS row stride (bf16 elems) -> conflict-free ds_load_b128
#define NK  ((3 * EMB_) / BK)

__global__ void __launch_bounds__(256)
gemm_layer(const unsigned short* __restrict__ Abf,   // 3 x [NN,512] bf16 concatenated along K
           const unsigned short* __restrict__ WbT,   // [512][1536] bf16 (N-major)
           const float* __restrict__ bgc, const float* __restrict__ bgc2,
           const float* __restrict__ bbi,
           float* __restrict__ zbuf) {
#if HAS_ASYNC_LDS
  __shared__ alignas(16) unsigned short lAs[2][BM * LDT];
  __shared__ alignas(16) unsigned short lBs[2][BN * LDT];
#else
  __shared__ alignas(16) unsigned short lAs[1][BM * LDT];
  __shared__ alignas(16) unsigned short lBs[1][BN * LDT];
#endif

  const int m0 = blockIdx.y * BM;
  const int n0 = blockIdx.x * BN;

  const int tid  = threadIdx.x;
  const int lane = tid & 31;
  const int wid  = tid >> 5;
  const int wm   = wid & 3;        // M sub-tile (32 rows)
  const int wn   = wid >> 2;       // N sub-tile (64 cols)
  const int l15  = lane & 15;
  const int lh   = lane >> 4;

  const int tr = tid >> 2;         // loader row 0..63 (and +64)
  const int tc = (tid & 3) * 8;    // 16B chunk offset (bf16 elems)

  // row clamp: output row m depends only on A row m; rows >= NN_ are never
  // stored, so clamping (instead of zero-fill) is safe and branch-free.
  int grA[2];
#pragma unroll
  for (int h = 0; h < 2; ++h) {
    int gr = m0 + tr + h * 64;
    grA[h] = (gr < NN_) ? gr : (NN_ - 1);
  }

  v8f accv[2][4] = {};

#if HAS_ASYNC_LDS
  // ---- async double-buffered tile pipeline (ASYNCcnt) ----
#define ISSUE_TILE(KT, BUF)                                                          \
  do {                                                                               \
    const int kcat_ = (KT) * BK;                                                     \
    const int seg_  = kcat_ >> 9;                                                    \
    const size_t ab_ = (size_t)kcat_ + (size_t)seg_ * ((size_t)EMB_ * (NN_ - 1));    \
    _Pragma("unroll")                                                                \
    for (int h = 0; h < 2; ++h) {                                                    \
      const int r_ = tr + h * 64;                                                    \
      async_ld128(Abf + ab_ + (size_t)grA[h] * EMB_ + tc, &lAs[BUF][r_ * LDT + tc]); \
      async_ld128(WbT + (size_t)(n0 + r_) * (3 * EMB_) + kcat_ + tc,                 \
                  &lBs[BUF][r_ * LDT + tc]);                                         \
    }                                                                                \
  } while (0)

  ISSUE_TILE(0, 0);
  for (int kt = 0; kt < NK; ++kt) {
    const int cur = kt & 1;
    wait_async0();            // my 4 async loads of tile kt have landed in LDS
    __syncthreads();          // everyone's tile-kt loads done; everyone done computing kt-1
    if (kt + 1 < NK)
      ISSUE_TILE(kt + 1, cur ^ 1);   // overlaps with compute below

    FragU fa[2], fb[4];
#pragma unroll
    for (int ti = 0; ti < 2; ++ti) {
      const int ab = (wm * 32 + ti * 16 + l15) * LDT + lh * 8;   // K0-7 / K8-15
      fa[ti].q[0] = *(const uint4*)&lAs[cur][ab];
      fa[ti].q[1] = *(const uint4*)&lAs[cur][ab + 16];           // K16-23 / K24-31
    }
#pragma unroll
    for (int tj = 0; tj < 4; ++tj) {
      const int bb = (wn * 64 + tj * 16 + l15) * LDT + lh * 8;   // K0-7 / K8-15, then K16-23 / K24-31
      fb[tj].q[0] = *(const uint4*)&lBs[cur][bb];
      fb[tj].q[1] = *(const uint4*)&lBs[cur][bb + 16];
    }
#pragma unroll
    for (int ti = 0; ti < 2; ++ti)
#pragma unroll
      for (int tj = 0; tj < 4; ++tj)
        accv[ti][tj] = __builtin_amdgcn_wmma_f32_16x16x32_bf16(
            false, fa[ti].v, false, fb[tj].v, (short)0, accv[ti][tj], false, false);
  }
#undef ISSUE_TILE
#else
  // ---- synchronous fallback (branch-free clamped loads) ----
  for (int kt = 0; kt < NK; ++kt) {
    const int kcat = kt * BK;
    const int seg  = kcat >> 9;
    const size_t ab = (size_t)kcat + (size_t)seg * ((size_t)EMB_ * (NN_ - 1));

    __syncthreads();
#pragma unroll
    for (int h = 0; h < 2; ++h) {
      const int r = tr + h * 64;
      const uint4 av = *(const uint4*)(Abf + ab + (size_t)grA[h] * EMB_ + tc);
      *(uint4*)&lAs[0][r * LDT + tc] = av;
      const uint4 bv = *(const uint4*)(WbT + (size_t)(n0 + r) * (3 * EMB_) + kcat + tc);
      *(uint4*)&lBs[0][r * LDT + tc] = bv;
    }
    __syncthreads();

    FragU fa[2], fb[4];
#pragma unroll
    for (int ti = 0; ti < 2; ++ti) {
      const int ab2 = (wm * 32 + ti * 16 + l15) * LDT + lh * 8;
      fa[ti].q[0] = *(const uint4*)&lAs[0][ab2];
      fa[ti].q[1] = *(const uint4*)&lAs[0][ab2 + 16];
    }
#pragma unroll
    for (int tj = 0; tj < 4; ++tj) {
      const int bb = (wn * 64 + tj * 16 + l15) * LDT + lh * 8;
      fb[tj].q[0] = *(const uint4*)&lBs[0][bb];
      fb[tj].q[1] = *(const uint4*)&lBs[0][bb + 16];
    }
#pragma unroll
    for (int ti = 0; ti < 2; ++ti)
#pragma unroll
      for (int tj = 0; tj < 4; ++tj)
        accv[ti][tj] = __builtin_amdgcn_wmma_f32_16x16x32_bf16(
            false, fa[ti].v, false, fb[tj].v, (short)0, accv[ti][tj], false, false);
  }
#endif

  // fused bias epilogue, f32 store
#pragma unroll
  for (int tj = 0; tj < 4; ++tj) {
    const int col = n0 + wn * 64 + tj * 16 + l15;
    const float bias = bgc[col] + bgc2[col] + bbi[col];
#pragma unroll
    for (int ti = 0; ti < 2; ++ti) {
      const int rbase = m0 + wm * 32 + ti * 16 + lh * 8;
#pragma unroll
      for (int j = 0; j < 8; ++j) {
        const int row = rbase + j;
        if (row < NN_)
          zbuf[(size_t)row * EMB_ + col] = accv[ti][tj][j] + bias;
      }
    }
  }
}

// ---------------------------------------------------------------- leaky-relu + L2 row-normalize + accumulate (wave per row)
__global__ void epilogue_norm(const float* __restrict__ z, float* __restrict__ ego,
                              float* __restrict__ acc) {
  const int row = blockIdx.x * 8 + (threadIdx.x >> 5);
  if (row >= NN_) return;
  const int l = threadIdx.x & 31;
  const float4* zr = (const float4*)(z + (size_t)row * EMB_);
  float4* er = (float4*)(ego + (size_t)row * EMB_);
  float4* ar = (float4*)(acc + (size_t)row * EMB_);
  float4 e[4];
  float ss = 0.f;
#pragma unroll
  for (int j = 0; j < 4; ++j) {
    float4 v = zr[l + j * 32];
    v.x = (v.x >= 0.f) ? v.x : LRELU_ * v.x;
    v.y = (v.y >= 0.f) ? v.y : LRELU_ * v.y;
    v.z = (v.z >= 0.f) ? v.z : LRELU_ * v.z;
    v.w = (v.w >= 0.f) ? v.w : LRELU_ * v.w;
    e[j] = v;
    ss += v.x * v.x + v.y * v.y + v.z * v.z + v.w * v.w;
  }
#pragma unroll
  for (int off = 16; off > 0; off >>= 1)
    ss += __shfl_xor(ss, off);
  const float sc = 1.f / fmaxf(sqrtf(ss), EPS_);
#pragma unroll
  for (int j = 0; j < 4; ++j) {
    er[l + j * 32] = e[j];
    float4 a = ar[l + j * 32];
    a.x += e[j].x * sc; a.y += e[j].y * sc; a.z += e[j].z * sc; a.w += e[j].w * sc;
    ar[l + j * 32] = a;
  }
}

// ---------------------------------------------------------------- final gather: (v_emb, t_emb)
__global__ void gather_out(const float* __restrict__ acc, const int* __restrict__ items,
                           float* __restrict__ out) {
  const int b = blockIdx.x;
  const int t = threadIdx.x;    // 128 threads -> float4 per thread
  const int it = items[b];
  const float4 v  = ((const float4*)(acc + (size_t)it * EMB_))[t];
  const float4 tt = ((const float4*)(acc + (size_t)(N_ITEM_ + it) * EMB_))[t];
  ((float4*)out)[(size_t)b * 128 + t] = v;
  ((float4*)(out + (size_t)BATCH_ * EMB_))[(size_t)b * 128 + t] = tt;
}

// ----------------------------------------------------------------
extern "C" void kernel_launch(void* const* d_in, const int* in_sizes, int n_in,
                              void* d_out, int out_size, void* d_ws, size_t ws_size,
                              hipStream_t stream) {
  (void)in_sizes; (void)n_in; (void)out_size; (void)ws_size;
  const float* vis   = (const float*)d_in[0];
  const float* txt   = (const float*)d_in[1];
  const int*   erow  = (const int*)d_in[2];
  const int*   ecol  = (const int*)d_in[3];
  const float* eval  = (const float*)d_in[4];
  const int*   items = (const int*)d_in[5];
  const float* Wgc   = (const float*)d_in[6];
  const float* bgc   = (const float*)d_in[7];
  const float* Wgc2  = (const float*)d_in[8];
  const float* bgc2  = (const float*)d_in[9];
  const float* Wbi   = (const float*)d_in[10];
  const float* bbi   = (const float*)d_in[11];
  float* out = (float*)d_out;

  const size_t MAT = (size_t)NN_ * EMB_;           // 51.2M elems
  float* ego  = (float*)d_ws;
  float* acc  = ego  + MAT;
  float* edis = acc  + MAT;
  float* side = edis + MAT;
  float* zbuf = side;                              // alias: side dead once Abf is packed
  unsigned short* Abf = (unsigned short*)(side + MAT);   // 3 x MAT bf16
  unsigned short* WbT = Abf + 3 * MAT;                   // 512*1536 bf16

  const int nBlkInit = (int)((MAT / 4 + 255) / 256);
  init_ego_acc<<<nBlkInit, 256, 0, stream>>>(vis, txt, ego, acc);

  // edis = spmm(oge) is constant across layers: compute once
  (void)hipMemsetAsync(edis, 0, MAT * sizeof(float), stream);
  spmm_atomic<<<(NNZ_ + 7) / 8, 256, 0, stream>>>(erow, ecol, eval, txt, vis, edis);

  const dim3 ggrid(EMB_ / BN, (NN_ + BM - 1) / BM);
  for (int k = 0; k < LAYERS_; ++k) {
    (void)hipMemsetAsync(side, 0, MAT * sizeof(float), stream);
    spmm_atomic<<<(NNZ_ + 7) / 8, 256, 0, stream>>>(
        erow, ecol, eval, ego, ego + (size_t)N_ITEM_ * EMB_, side);
    build_abf<<<(int)((MAT / 2 + 255) / 256), 256, 0, stream>>>(
        side, edis, ego, vis, txt,
        (unsigned*)Abf, (unsigned*)(Abf + MAT), (unsigned*)(Abf + 2 * MAT));
    convert_weights<<<(3 * EMB_ * EMB_) / 256, 256, 0, stream>>>(
        Wgc + (size_t)k * EMB_ * EMB_, Wgc2 + (size_t)k * EMB_ * EMB_,
        Wbi + (size_t)k * EMB_ * EMB_, WbT);
    gemm_layer<<<ggrid, 256, 0, stream>>>(
        Abf, WbT, bgc + k * EMB_, bgc2 + k * EMB_, bbi + k * EMB_, zbuf);
    epilogue_norm<<<(NN_ + 7) / 8, 256, 0, stream>>>(zbuf, ego, acc);
  }
  gather_out<<<BATCH_, 128, 0, stream>>>(acc, items, out);
}
